// ConvLSTM_Model_4836133175920
// MI455X (gfx1250) — matrix operand order repaired and checked by reading, not compiled, not run on hardware
//
#include <hip/hip_runtime.h>
#include <hip/hip_bf16.h>

// ---------------------------------------------------------------------------
// ConvLSTM2D(F=8, 5x5, SAME, relu) + inference BatchNorm for MI455X (gfx1250).
//
// One persistent 1024-thread workgroup per batch element (grid = 32), looping
// over T=25 with workgroup barriers. All recurrent state lives in LDS:
//   h: double-buffered f16, halo-padded 68x69x8  (2 x 73 KB, extra zero row)
//   c: f32 64x64x8                                (128 KB)
//   x_t: f16 halo-padded 68x70                    (9.3 KB, extra zero rows)
//   Wh/Wx packed as per-lane WMMA B fragments     (16 KB)
// Fused input+recurrent conv as one im2col GEMM per 16-pixel tile:
//   M = 16 pixels, N = 32 gate channels (2 n-tiles),
//   K = [32 x-taps][7 x 32 h-(tap,cin)] -> v_wmma_f32_16x16x32_f16,
// f32 accumulate, bias folded into the initial accumulator.
// Two same-row tiles are processed concurrently (4 independent acc chains);
// all h-im2col loads are ds_load_b128 from two per-tile base pointers
// (hb + hi*16 / hb + hi*1024) with compile-time immediate offsets, so no
// offset registers stay live (no scratch spills). Out-of-kernel taps hit
// zeroed padding rows (weights there are zero; data must stay finite).
// ---------------------------------------------------------------------------

typedef __attribute__((ext_vector_type(16))) _Float16 v16h;
typedef __attribute__((ext_vector_type(8)))  float    v8f;

#define HH   64
#define WW   64
#define PW   68                    // padded width (2-halo each side)
#define PHR  69                    // padded h rows (2-halo + 1 extra zero row)
#define PXR  70                    // padded x rows (2-halo + 2 extra zero rows)
#define FF   8
#define TT   25
#define NPIX (HH * WW)             // 4096
#define NTHREADS 1024

#define HBUF_HALVES (PW * PHR * FF)   // 37536
#define XBUF_HALVES (PW * PXR)        // 4760

// LDS layout (bytes)
#define OFF_H0   0                 // 75072 B
#define OFF_H1   75072             // 75072 B
#define OFF_C    150144            // 131072 B
#define OFF_X    281216            // 9520 B
#define OFF_WB   290736            // 16384 B packed B frags
#define OFF_BN   307120            // 64 B
#define SMEM_BYTES 307184

union FragU {
  v16h     v;
  _Float16 h[16];
  uint4    q[2];
};

__device__ __forceinline__ float hsig(float z) {
  return fminf(fmaxf(0.2f * z + 0.5f, 0.0f), 1.0f);
}

// compile-time im2col offsets
__host__ __device__ constexpr int tapOffH(int tap) {        // bytes into h
  return ((tap / 5) * PW + (tap % 5)) * (FF * 2);
}
__host__ __device__ constexpr int tapOffX(int K) {          // elements into x
  return (K / 5) * PW + (K % 5);
}
__host__ __device__ constexpr int kmap(int hh) {            // A-frag half -> K
  return (hh < 8) ? hh : hh + 8;
}

__global__ __launch_bounds__(NTHREADS, 1)
void convlstm2d_bn_kernel(const float* __restrict__ xg,
                          const float* __restrict__ Wx,   // (5,5,1,32) flat tap*32+ch
                          const float* __restrict__ Wh,   // (5,5,8,32) flat (tap*8+cin)*32+ch
                          const float* __restrict__ bias, // (32)
                          const float* __restrict__ gamma,
                          const float* __restrict__ beta,
                          const float* __restrict__ mean,
                          const float* __restrict__ var,
                          float* __restrict__ out) {
  extern __shared__ char smem[];
  _Float16* hbuf0 = (_Float16*)(smem + OFF_H0);
  _Float16* hbuf1 = (_Float16*)(smem + OFF_H1);
  float*    cbuf  = (float*)   (smem + OFF_C);
  _Float16* xbuf  = (_Float16*)(smem + OFF_X);
  _Float16* wB    = (_Float16*)(smem + OFF_WB);
  float*    bnA   = (float*)   (smem + OFF_BN);
  float*    bnB   = bnA + 8;

  const int tid  = threadIdx.x;
  const int lane = tid & 31;
  const int wave = tid >> 5;
  const int n    = lane & 15;   // D-frag column / A-frag row (pixel m)
  const int hi   = lane >> 4;   // half-wave selector
  const int b    = blockIdx.x;

  // ---- init: zero padded h buffers, padded x, c ---------------------------
  for (int i = tid; i < HBUF_HALVES; i += NTHREADS) {
    hbuf0[i] = (_Float16)0.0f;
    hbuf1[i] = (_Float16)0.0f;
  }
  for (int i = tid; i < XBUF_HALVES; i += NTHREADS) xbuf[i] = (_Float16)0.0f;
#pragma unroll
  for (int i = 0; i < NPIX * FF / NTHREADS; ++i) cbuf[tid + i * NTHREADS] = 0.0f;

  // ---- pack B fragments: half index f = ((ck*2+nt)*32 + lane)*16 + hh -----
  //   B element: K = (hh/8)*16 + (lane/16)*8 + hh%8, N = nt*16 + lane%16
#pragma unroll
  for (int i = 0; i < 8; ++i) {
    int f  = tid * 8 + i;
    int hh = f & 15;
    int ln = (f >> 4) & 31;
    int nt = (f >> 9) & 1;
    int ck = f >> 10;
    int K  = ((hh >> 3) << 4) + ((ln >> 4) << 3) + (hh & 7);
    int N  = nt * 16 + (ln & 15);
    float v = 0.0f;
    if (ck == 0) {
      if (K < 25) v = Wx[K * 32 + N];
    } else {
      int kp = (ck - 1) * 32 + K;     // kp = tap*8 + cin
      if (kp < 200) v = Wh[kp * 32 + N];
    }
    wB[f] = (_Float16)v;
  }

  // ---- fold BN into y = h*bnA + bnB ---------------------------------------
  if (tid < 8) {
    float a = rsqrtf(var[tid] + 1e-3f) * gamma[tid];
    bnA[tid] = a;
    bnB[tid] = beta[tid] - mean[tid] * a;
  }

  const float bias0 = bias[n];
  const float bias1 = bias[n + 16];
  const int   hi16   = hi << 4;    // hi-step when dx(tap0) < 4
  const int   hi1024 = hi << 10;   // hi-step when dx(tap0) == 4 ((PW-4)*16)

  __syncthreads();

  const char* wBlane = (const char*)wB + lane * 32;  // per-lane B frag base

  for (int t = 0; t < TT; ++t) {
    // stage x[b,t] into padded LDS tile as f16 (halo/extra rows stay zero)
    const float* xt = xg + ((size_t)b * TT + t) * NPIX;
#pragma unroll
    for (int i = 0; i < NPIX / NTHREADS; ++i) {
      int p = tid + i * NTHREADS;
      xbuf[((p >> 6) + 2) * PW + (p & 63) + 2] = (_Float16)xt[p];
    }
    if (t + 1 < TT)
      __builtin_prefetch(xt + NPIX + tid, 0, 0);  // global_prefetch_b8
    __syncthreads();

    const _Float16* hcur = (t & 1) ? hbuf1 : hbuf0;
    _Float16*       hnxt = (t & 1) ? hbuf0 : hbuf1;
    const size_t obase = ((size_t)b * TT + t) * (size_t)(NPIX * FF);

    // wave owns tiles [wave*8, wave*8+8): 4 pairs of same-row tiles
    for (int pr = 0; pr < 4; ++pr) {
      const int tile = (wave << 3) + (pr << 1);   // tile and tile+1, same row
      const int ty   = tile >> 2;
      const int x0   = (tile & 3) << 4;           // 0 or 32
      const int tb   = ty * PW + x0 + n;          // padded-coord im2col base

      v8f acc0[2], acc1[2];
#pragma unroll
      for (int r = 0; r < 8; ++r) {
        acc0[0][r] = bias0; acc0[1][r] = bias0;
        acc1[0][r] = bias1; acc1[1][r] = bias1;
      }

      FragU A[2], B0, B1;

      // -- chunk 0: input-conv im2col from padded x (branch-free) -----------
      // per-half element offset: hi ? tapOffX(K+8) : tapOffX(K), compile-time
      const _Float16* xb = xbuf + tb;
#pragma unroll
      for (int hh = 0; hh < 16; ++hh) {
        int off = hi ? tapOffX(kmap(hh) + 8) : tapOffX(kmap(hh));
        const _Float16* p = xb + off;
        A[0].h[hh] = p[0];
        A[1].h[hh] = p[16];
      }
      {
        const uint4* w0 = (const uint4*)(wBlane + 0 * 1024);
        const uint4* w1 = (const uint4*)(wBlane + 1 * 1024);
        B0.q[0] = w0[0]; B0.q[1] = w0[1];
        B1.q[0] = w1[0]; B1.q[1] = w1[1];
        acc0[0] = __builtin_amdgcn_wmma_f32_16x16x32_f16(false, A[0].v, false, B0.v,
                                                         (short)0, acc0[0], false, false);
        acc0[1] = __builtin_amdgcn_wmma_f32_16x16x32_f16(false, A[1].v, false, B0.v,
                                                         (short)0, acc0[1], false, false);
        acc1[0] = __builtin_amdgcn_wmma_f32_16x16x32_f16(false, A[0].v, false, B1.v,
                                                         (short)0, acc1[0], false, false);
        acc1[1] = __builtin_amdgcn_wmma_f32_16x16x32_f16(false, A[1].v, false, B1.v,
                                                         (short)0, acc1[1], false, false);
      }

      // -- chunks 1..7: recurrent-conv im2col from padded h (branch-free) ---
      // lane tap = tap0 + hi; hi-step is +16B (dx(tap0)<4) or +1024B (dx==4);
      // constant part folds into the ds immediate offset. tile+1: +256B.
      const char* hb  = (const char*)hcur + tb * (FF * 2);
      const char* hbA = hb + hi16;
      const char* hbB = hb + hi1024;
#pragma unroll
      for (int ck = 1; ck < 8; ++ck) {
#pragma unroll
        for (int j = 0; j < 2; ++j) {
          const int tap0 = (ck - 1) * 4 + 2 * j;
          const char* base = ((tap0 % 5) == 4) ? hbB : hbA;
          A[0].q[j] = *(const uint4*)(base + tapOffH(tap0));
          A[1].q[j] = *(const uint4*)(base + tapOffH(tap0) + 256);
        }
        const uint4* w0 = (const uint4*)(wBlane + (ck * 2 + 0) * 1024);
        const uint4* w1 = (const uint4*)(wBlane + (ck * 2 + 1) * 1024);
        B0.q[0] = w0[0]; B0.q[1] = w0[1];
        B1.q[0] = w1[0]; B1.q[1] = w1[1];
        acc0[0] = __builtin_amdgcn_wmma_f32_16x16x32_f16(false, A[0].v, false, B0.v,
                                                         (short)0, acc0[0], false, false);
        acc0[1] = __builtin_amdgcn_wmma_f32_16x16x32_f16(false, A[1].v, false, B0.v,
                                                         (short)0, acc0[1], false, false);
        acc1[0] = __builtin_amdgcn_wmma_f32_16x16x32_f16(false, A[0].v, false, B1.v,
                                                         (short)0, acc1[0], false, false);
        acc1[1] = __builtin_amdgcn_wmma_f32_16x16x32_f16(false, A[1].v, false, B1.v,
                                                         (short)0, acc1[1], false, false);
      }

      // -- gate recombination + LSTM cell update for both tiles -------------
#pragma unroll
      for (int u = 0; u < 2; ++u) {
        float fo0[8], fo1[8];
#pragma unroll
        for (int r = 0; r < 8; ++r) {
          fo0[r] = __shfl_xor(acc0[u][r], 8, 32);
          fo1[r] = __shfl_xor(acc1[u][r], 8, 32);
        }

        if (n < 8) {             // 16 active lanes x 8 cols x cin=n
          const int cin = n;
          const float a  = bnA[cin];
          const float bb = bnB[cin];
          const int xu = x0 + 16 * u;
#pragma unroll
          for (int r = 0; r < 8; ++r) {
            int M   = r + hi * 8;               // pixel column within tile
            int col = xu + M;
            int px  = ty * WW + col;            // unpadded pixel index
            int hp  = ((ty + 2) * PW + col + 2) * FF + cin;  // padded h index
            float zi = acc0[u][r], zg = acc1[u][r], zf = fo0[r], zo = fo1[r];
            float ig = hsig(zi);
            float fg = hsig(zf);
            float og = hsig(zo);
            float g  = fmaxf(zg, 0.0f);         // activation = relu
            int ci = px * FF + cin;
            float cn = fg * cbuf[ci] + ig * g;
            cbuf[ci] = cn;
            float hn = og * fmaxf(cn, 0.0f);
            hnxt[hp] = (_Float16)hn;
            out[obase + (size_t)px * FF + cin] = hn * a + bb;
          }
        }
      }
    }
    __syncthreads();   // all h_next written / h_cur reads done before next t
  }
}

extern "C" void kernel_launch(void* const* d_in, const int* in_sizes, int n_in,
                              void* d_out, int out_size, void* d_ws, size_t ws_size,
                              hipStream_t stream) {
  (void)in_sizes; (void)n_in; (void)out_size; (void)d_ws; (void)ws_size;
  const float* x     = (const float*)d_in[0];
  const float* Wx    = (const float*)d_in[1];
  const float* Wh    = (const float*)d_in[2];
  const float* bias  = (const float*)d_in[3];
  const float* gamma = (const float*)d_in[4];
  const float* beta  = (const float*)d_in[5];
  const float* mean  = (const float*)d_in[6];
  const float* var   = (const float*)d_in[7];
  float* out = (float*)d_out;

  // Allow the large dynamic LDS allocation (WGP has 320KB on CDNA5).
  hipFuncSetAttribute((const void*)convlstm2d_bn_kernel,
                      hipFuncAttributeMaxDynamicSharedMemorySize, SMEM_BYTES);

  convlstm2d_bn_kernel<<<dim3(32), dim3(NTHREADS), SMEM_BYTES, stream>>>(
      x, Wx, Wh, bias, gamma, beta, mean, var, out);
}
